// StarClassifierFusionTransformer_44367012167908
// MI455X (gfx1250) — compile-verified, matrix operand order checked by reading, not run on hardware
//
#include <hip/hip_runtime.h>
#include <hip/hip_bf16.h>

// ---------------------------------------------------------------------------
// MI455X / gfx1250 implementation. All matmul-heavy work runs on the WMMA
// pipe (v_wmma_f32_16x16x32_bf16, wave32). fp32 master residuals, bf16
// operands for every GEMM / attention matmul. Memory-bound workload
// (~1.1 TFLOP vs ~300MB traffic @ 23.3 TB/s): operand bytes halved, all hot
// loads are 16B global_load_b128 / ds_load_b128 patterns, and the weight
// fp32->bf16 transpose is LDS-tiled so the dominant HBM stream (weights)
// stays fully coalesced. K-loop kept branch-free so the scheduler pipelines
// global_load_b128 clauses ahead of the WMMAs (s_wait_loadcnt > 0).
// ---------------------------------------------------------------------------

#define BZ   64
#define DM   512
#define NL   6
#define NH   8
#define HD   64
#define SS   256          // spectra tokens
#define SG   18           // gaia tokens
#define MS   (BZ*SS)      // 16384
#define MG   (BZ*SG)      // 1152
#define NC   55

typedef __attribute__((ext_vector_type(8)))  __bf16 v8bf;
typedef __attribute__((ext_vector_type(16))) __bf16 v16bf;
typedef __attribute__((ext_vector_type(8)))  float  v8f;
typedef unsigned short ushort;

__device__ __forceinline__ ushort f2bf(float f) {
    unsigned u = __float_as_uint(f);
    u = (u + 0x7FFFu + ((u >> 16) & 1u)) >> 16;
    return (ushort)u;
}

// Load one A/B operand fragment for wmma_f32_16x16x32_bf16.
// rowp points at the (row-major, contiguous-K) row for this lane.
// Elements 0..7  = K[k0 + half*8 .. +7], elements 8..15 = K[k0+16+half*8 .. +7]
__device__ __forceinline__ v16bf load_ab(const ushort* rowp, int k0, int half) {
    v8bf lo = *(const v8bf*)(rowp + k0 + half * 8);
    v8bf hi = *(const v8bf*)(rowp + k0 + 16 + half * 8);
    return __builtin_shufflevector(lo, hi, 0,1,2,3,4,5,6,7,8,9,10,11,12,13,14,15);
}

__device__ __forceinline__ v8f wmma_bf16(v16bf a, v16bf b, v8f c) {
    return __builtin_amdgcn_wmma_f32_16x16x32_bf16(false, a, false, b, (short)0, c, false, false);
}

// ---------------------------------------------------------------------------
// Weight transpose + fp32->bf16: Wt[n*K + k] = bf16(W[k*ld + coloff + n]).
// LDS-tiled 32x32 so both the fp32 reads (128B/row) and bf16 writes (64B/row)
// are coalesced; odd-padded tile avoids LDS bank conflicts on the transposed
// read-back. Weights are the dominant HBM stream -> this must not amplify.
// ---------------------------------------------------------------------------
__global__ __launch_bounds__(256) void transpose_bf16_kernel(
    const float* __restrict__ W, ushort* __restrict__ Wt,
    int K, int N, int ld, int coloff) {
    __shared__ ushort tile[32][33];
    int tilesK = K >> 5;
    int tk = blockIdx.x % tilesK;
    int tn = blockIdx.x / tilesK;
    int tx = threadIdx.x & 31;
    int ty = threadIdx.x >> 5;          // 0..7
    int k0 = tk * 32, n0 = tn * 32;
    #pragma unroll
    for (int j = 0; j < 4; ++j) {
        int k = k0 + ty + j * 8;
        tile[ty + j * 8][tx] = f2bf(W[(size_t)k * ld + coloff + n0 + tx]);
    }
    __syncthreads();
    #pragma unroll
    for (int j = 0; j < 4; ++j) {
        int n = n0 + ty + j * 8;
        Wt[(size_t)n * K + k0 + tx] = tile[tx][ty + j * 8];
    }
}

__global__ void f2bf_kernel(const float* __restrict__ in, ushort* __restrict__ out, int n) {
    int idx = blockIdx.x * 256 + threadIdx.x;
    if (idx < n) out[idx] = f2bf(in[idx]);
}

__global__ void zero_bf16_kernel(ushort* __restrict__ p, int n) {
    int idx = blockIdx.x * 256 + threadIdx.x;
    if (idx < n) p[idx] = 0;
}

// ---------------------------------------------------------------------------
// Tokenizer: out[m*N+n] = sum_k xin[m*Ktok+k]*w[k*N+n] + b[n]   (K=64 or 2)
// ---------------------------------------------------------------------------
__global__ void token_kernel(const float* __restrict__ xin, const float* __restrict__ w,
                             const float* __restrict__ bias, float* __restrict__ out,
                             int Ktok, int N, int total) {
    int idx = blockIdx.x * 256 + threadIdx.x;
    if (idx >= total) return;
    int m = idx / N, n = idx % N;
    float a = bias[n];
    for (int k = 0; k < Ktok; ++k)
        a += xin[(size_t)m * Ktok + k] * w[(size_t)k * N + n];
    out[idx] = a;
}

// ---------------------------------------------------------------------------
// LayerNorm (per row of Dm fp32) -> bf16 row-major
// ---------------------------------------------------------------------------
__global__ __launch_bounds__(256) void layernorm_bf16_kernel(
    const float* __restrict__ x, const float* __restrict__ w, const float* __restrict__ b,
    ushort* __restrict__ out, int Dm) {
    __shared__ float red[256];
    int t = threadIdx.x;
    const float* row = x + (size_t)blockIdx.x * Dm;
    float s0 = 0.f, s1 = 0.f;
    for (int d = t; d < Dm; d += 256) { float v = row[d]; s0 += v; s1 += v * v; }
    red[t] = s0; __syncthreads();
    for (int off = 128; off; off >>= 1) { if (t < off) red[t] += red[t + off]; __syncthreads(); }
    float mean = red[0] / (float)Dm; __syncthreads();
    red[t] = s1; __syncthreads();
    for (int off = 128; off; off >>= 1) { if (t < off) red[t] += red[t + off]; __syncthreads(); }
    float var = red[0] / (float)Dm - mean * mean;
    float inv = rsqrtf(var + 1e-5f);
    ushort* orow = out + (size_t)blockIdx.x * Dm;
    for (int d = t; d < Dm; d += 256)
        orow[d] = f2bf((row[d] - mean) * inv * w[d] + b[d]);
}

// ---------------------------------------------------------------------------
// WMMA GEMM: C(MxN) = A(bf16 MxK, row-major) * Bt(bf16 NxK)^T + bias
//  mode 0: fp32 row-major out, + residual if res!=nullptr
//  mode 1: bf16 out to (B,H,S,HD); optional fused RoPE (pair hd+-32 is in
//          acc[sub^2] of the same lane because one wave-tile == one head)
//  mode 2: bf16 out to (B,H,HD,Sp)  (V transposed, key-dim padded to Sp)
//  mode 3: exact GELU, bf16 row-major out
// One wave computes a 16x64 tile (4 WMMA accumulators), K-loop step 32.
// The 8 waves of a block share the same 16-row A panel (consecutive gw =
// consecutive tn), so A hits WGP$/L2 after the first wave touches it.
// NOTE: keep the K-loop a single basic block (no branches/prefetch) so the
// backend software-pipelines the load clauses across the WMMAs.
// ---------------------------------------------------------------------------
__global__ __launch_bounds__(256) void gemm_wmma_kernel(
    const ushort* __restrict__ A, const ushort* __restrict__ Bt,
    const float* __restrict__ bias, const float* __restrict__ res,
    float* __restrict__ outf, ushort* __restrict__ outh,
    int M, int N, int K, int mode, int rope, int S, int Sp) {
    int wid = threadIdx.x >> 5;
    int gw = blockIdx.x * 8 + wid;
    int tilesN = N >> 6;
    if (gw >= (M >> 4) * tilesN) return;
    int tm = gw / tilesN, tn = gw % tilesN;
    int lane = threadIdx.x & 31, half = lane >> 4, nl = lane & 15;

    const ushort* arow = A + (size_t)(tm * 16 + nl) * K;
    const ushort* brow0 = Bt + (size_t)(tn * 64 + 0 * 16 + nl) * K;
    const ushort* brow1 = Bt + (size_t)(tn * 64 + 1 * 16 + nl) * K;
    const ushort* brow2 = Bt + (size_t)(tn * 64 + 2 * 16 + nl) * K;
    const ushort* brow3 = Bt + (size_t)(tn * 64 + 3 * 16 + nl) * K;

    v8f acc[4] = {};
    for (int k0 = 0; k0 < K; k0 += 32) {
        v16bf a = load_ab(arow, k0, half);
        acc[0] = wmma_bf16(a, load_ab(brow0, k0, half), acc[0]);
        acc[1] = wmma_bf16(a, load_ab(brow1, k0, half), acc[1]);
        acc[2] = wmma_bf16(a, load_ab(brow2, k0, half), acc[2]);
        acc[3] = wmma_bf16(a, load_ab(brow3, k0, half), acc[3]);
    }

    #pragma unroll
    for (int sub = 0; sub < 4; ++sub) {
        #pragma unroll
        for (int r = 0; r < 8; ++r) {
            int m = tm * 16 + r + half * 8;
            int n = tn * 64 + sub * 16 + nl;
            float v = acc[sub][r] + bias[n];
            if (mode == 0) {
                size_t idx = (size_t)m * N + n;
                if (res) v += res[idx];
                outf[idx] = v;
            } else if (mode == 1) {
                if (rope) {
                    int np = tn * 64 + (sub ^ 2) * 16 + nl;
                    float pv = acc[sub ^ 2][r] + bias[np];
                    int hd = sub * 16 + nl;            // 0..63 (one head per tile)
                    int d2 = hd & 31;
                    float ang = (float)(m % S) * expf(-(float)d2 * 0.28782313662425574f);
                    float c = cosf(ang), sn = sinf(ang);
                    v = v * c + ((hd < 32) ? -pv : pv) * sn;
                }
                int bb = m / S, s = m % S;
                int h = n >> 6, hd = n & 63;
                outh[(((size_t)bb * NH + h) * S + s) * HD + hd] = f2bf(v);
            } else if (mode == 2) {
                int bb = m / S, s = m % S;
                int h = n >> 6, hd = n & 63;
                outh[(((size_t)bb * NH + h) * HD + hd) * Sp + s] = f2bf(v);
            } else { // mode 3: exact GELU
                float g = 0.5f * v * (1.0f + erff(v * 0.70710678118654752f));
                outh[(size_t)m * N + n] = f2bf(g);
            }
        }
    }
}

// ---------------------------------------------------------------------------
// Attention: one wave per (b, h, 16-query tile).
//  q,k : bf16 (B,H,Sq,HD) / (B,H,Sk,HD) ; vt : bf16 (B,H,HD,Sp), Sp = pad32(Sk)
//  out : bf16 (B,Sq,DM) row-major (ready to be the next GEMM's A operand)
// QK^T and P*V both on v_wmma_f32_16x16x32_bf16; softmax fp32 through LDS.
// ---------------------------------------------------------------------------
__global__ __launch_bounds__(64) void attn_kernel(
    const ushort* __restrict__ qb, const ushort* __restrict__ kb,
    const ushort* __restrict__ vt, ushort* __restrict__ out,
    int Sq, int Sk, int Sp, int QT) {
    __shared__ float  sc[2][16][256];
    __shared__ ushort pp[2][16][256];
    int wid = threadIdx.x >> 5;
    int gwave = blockIdx.x * 2 + wid;
    int qt = gwave % QT;
    int rest = gwave / QT;
    int h = rest % NH;
    int b = rest / NH;
    int lane = threadIdx.x & 31, half = lane >> 4, nl = lane & 15;
    int q0 = qt * 16;

    // Q fragments (row = nl, clamped for padded query rows)
    int qrowi = q0 + nl; if (qrowi > Sq - 1) qrowi = Sq - 1;
    const ushort* qrow = qb + ((size_t)(b * NH + h) * Sq + qrowi) * HD;
    v16bf a0 = load_ab(qrow, 0, half);
    v16bf a1 = load_ab(qrow, 32, half);

    // scores = Q K^T / sqrt(HD)
    int KT = (Sk + 15) >> 4;
    for (int kt = 0; kt < KT; ++kt) {
        int key = kt * 16 + nl;
        int keyc = key > Sk - 1 ? Sk - 1 : key;
        const ushort* krow = kb + ((size_t)(b * NH + h) * Sk + keyc) * HD;
        v8f acc = {};
        acc = wmma_bf16(a0, load_ab(krow, 0, half), acc);
        acc = wmma_bf16(a1, load_ab(krow, 32, half), acc);
        #pragma unroll
        for (int r = 0; r < 8; ++r) {
            int mm = r + half * 8;
            float v = acc[r] * 0.125f;                 // 1/sqrt(64)
            if (key >= Sk) v = -__builtin_inff();
            sc[wid][mm][kt * 16 + nl] = v;
        }
    }
    __syncthreads();

    // softmax: lane L (<16) owns query row L of the tile
    if (lane < 16) {
        int row = lane;
        float mx = -__builtin_inff();
        for (int k = 0; k < Sk; ++k) mx = fmaxf(mx, sc[wid][row][k]);
        float sum = 0.f;
        for (int k = 0; k < Sk; ++k) { float e = expf(sc[wid][row][k] - mx); sum += e; sc[wid][row][k] = e; }
        float inv = 1.f / sum;
        for (int k = 0; k < Sp; ++k)
            pp[wid][row][k] = (k < Sk) ? f2bf(sc[wid][row][k] * inv) : (ushort)0;
    }
    __syncthreads();

    // O = P * V   (contraction over keys, padded to Sp)
    v8f oacc[4] = {};
    const ushort* prow = &pp[wid][nl][0];
    for (int kc = 0; kc < Sp; kc += 32) {
        v16bf pa = load_ab(prow, kc, half);
        #pragma unroll
        for (int sub = 0; sub < 4; ++sub) {
            const ushort* vrow = vt + ((size_t)(b * NH + h) * HD + sub * 16 + nl) * Sp;
            oacc[sub] = wmma_bf16(pa, load_ab(vrow, kc, half), oacc[sub]);
        }
    }
    #pragma unroll
    for (int sub = 0; sub < 4; ++sub) {
        #pragma unroll
        for (int r = 0; r < 8; ++r) {
            int s = q0 + r + half * 8;
            if (s < Sq) {
                int d = h * HD + sub * 16 + nl;
                out[((size_t)b * Sq + s) * DM + d] = f2bf(oacc[sub][r]);
            }
        }
    }
}

// ---------------------------------------------------------------------------
// Mean-pool both branches into fused (64 x 1024)
// ---------------------------------------------------------------------------
__global__ void pool_kernel(const float* __restrict__ xs, const float* __restrict__ xg,
                            float* __restrict__ fused) {
    int idx = blockIdx.x * 256 + threadIdx.x;
    if (idx >= BZ * 2 * DM) return;
    int b = idx >> 10, d = idx & 1023;
    float a = 0.f;
    if (d < DM) {
        for (int s = 0; s < SS; ++s) a += xs[((size_t)b * SS + s) * DM + d];
        a *= (1.f / SS);
    } else {
        int dd = d - DM;
        for (int s = 0; s < SG; ++s) a += xg[((size_t)b * SG + s) * DM + dd];
        a *= (1.f / SG);
    }
    fused[idx] = a;
}

// ---------------------------------------------------------------------------
// Final LayerNorm(1024) + classifier (1024 x 55). One block per batch row.
// ---------------------------------------------------------------------------
__global__ __launch_bounds__(256) void cls_kernel(
    const float* __restrict__ fused, const float* __restrict__ lnw, const float* __restrict__ lnb,
    const float* __restrict__ W, const float* __restrict__ bias, float* __restrict__ out) {
    __shared__ float hh[2 * DM];
    __shared__ float red[256];
    int b = blockIdx.x, t = threadIdx.x;
    const float* row = fused + (size_t)b * 2 * DM;
    float s0 = 0.f, s1 = 0.f;
    for (int d = t; d < 2 * DM; d += 256) { float v = row[d]; s0 += v; s1 += v * v; }
    red[t] = s0; __syncthreads();
    for (int off = 128; off; off >>= 1) { if (t < off) red[t] += red[t + off]; __syncthreads(); }
    float mean = red[0] / (2.f * DM); __syncthreads();
    red[t] = s1; __syncthreads();
    for (int off = 128; off; off >>= 1) { if (t < off) red[t] += red[t + off]; __syncthreads(); }
    float var = red[0] / (2.f * DM) - mean * mean;
    float inv = rsqrtf(var + 1e-5f);
    for (int d = t; d < 2 * DM; d += 256)
        hh[d] = (row[d] - mean) * inv * lnw[d] + lnb[d];
    __syncthreads();
    for (int c = t; c < NC; c += 256) {
        float a = bias[c];
        for (int d = 0; d < 2 * DM; ++d) a += hh[d] * W[(size_t)d * NC + c];
        out[(size_t)b * NC + c] = a;
    }
}

// ===========================================================================
// Host orchestration
// ===========================================================================
static inline int cdiv(int a, int b) { return (a + b - 1) / b; }

static void launch_gemm(hipStream_t st, const ushort* A, const ushort* Bt, const float* bias,
                        const float* res, float* outf, ushort* outh,
                        int M, int N, int K, int mode, int rope, int S, int Sp) {
    int waves = (M / 16) * (N / 64);
    gemm_wmma_kernel<<<cdiv(waves, 8), 256, 0, st>>>(A, Bt, bias, res, outf, outh,
                                                     M, N, K, mode, rope, S, Sp);
}

static void launch_transpose(hipStream_t st, const float* W, ushort* Wt,
                             int K, int N, int ld, int coloff) {
    transpose_bf16_kernel<<<(K / 32) * (N / 32), 256, 0, st>>>(W, Wt, K, N, ld, coloff);
}

struct Scratch {
    float *xs, *xg, *fused;
    ushort *lnb, *qb, *kb, *vtb, *ob, *hb, *ctxb, *wqkvT, *wT1, *wT2;
};

static void run_branch(hipStream_t st, float* x, int M, int S, int Sp,
                       const float* ln1w, const float* ln1b,
                       const float* qkv_w, const float* qkv_b,
                       const float* ow, const float* obias,
                       const float* ln2w, const float* ln2b,
                       const float* f1w, const float* f1b,
                       const float* f2w, const float* f2b,
                       const Scratch& sc) {
    int QT = cdiv(S, 16);
    for (int i = 0; i < NL; ++i) {
        // ln1
        layernorm_bf16_kernel<<<M, 256, 0, st>>>(x, ln1w + (size_t)i * DM, ln1b + (size_t)i * DM, sc.lnb, DM);
        // qkv weights -> bf16 N x K
        for (int j = 0; j < 3; ++j)
            launch_transpose(st, qkv_w + ((size_t)i * 3 + j) * DM * DM, sc.wqkvT + (size_t)j * DM * DM, DM, DM, DM, 0);
        const float* qb_bias = qkv_b + (size_t)i * 3 * DM;
        // Q, K with fused RoPE; V transposed (key-padded)
        launch_gemm(st, sc.lnb, sc.wqkvT + 0 * DM * DM, qb_bias + 0 * DM, nullptr, nullptr, sc.qb, M, DM, DM, 1, 1, S, Sp);
        launch_gemm(st, sc.lnb, sc.wqkvT + 1 * DM * DM, qb_bias + 1 * DM, nullptr, nullptr, sc.kb, M, DM, DM, 1, 1, S, Sp);
        if (Sp != S) {
            int nvt = BZ * NH * HD * Sp;
            zero_bf16_kernel<<<cdiv(nvt, 256), 256, 0, st>>>(sc.vtb, nvt);
        }
        launch_gemm(st, sc.lnb, sc.wqkvT + 2 * DM * DM, qb_bias + 2 * DM, nullptr, nullptr, sc.vtb, M, DM, DM, 2, 0, S, Sp);
        // attention
        attn_kernel<<<(BZ * NH * QT) / 2, 64, 0, st>>>(sc.qb, sc.kb, sc.vtb, sc.ob, S, S, Sp, QT);
        // O projection + residual (fp32 master)
        launch_transpose(st, ow + (size_t)i * DM * DM, sc.wT1, DM, DM, DM, 0);
        launch_gemm(st, sc.ob, sc.wT1, obias + (size_t)i * DM, x, x, nullptr, M, DM, DM, 0, 0, S, Sp);
        // ln2 + FFN
        layernorm_bf16_kernel<<<M, 256, 0, st>>>(x, ln2w + (size_t)i * DM, ln2b + (size_t)i * DM, sc.lnb, DM);
        launch_transpose(st, f1w + (size_t)i * DM * 4 * DM, sc.wT1, DM, 4 * DM, 4 * DM, 0);
        launch_gemm(st, sc.lnb, sc.wT1, f1b + (size_t)i * 4 * DM, nullptr, nullptr, sc.hb, M, 4 * DM, DM, 3, 0, S, Sp);
        launch_transpose(st, f2w + (size_t)i * 4 * DM * DM, sc.wT2, 4 * DM, DM, DM, 0);
        launch_gemm(st, sc.hb, sc.wT2, f2b + (size_t)i * DM, x, x, nullptr, M, DM, 4 * DM, 0, 0, S, Sp);
    }
}

static void run_cross(hipStream_t st, float* x, int Mq, int Sq,
                      const float* ctx, int Mk, int Sk, int Skp,
                      const float* lnw, const float* lnb1,
                      const float* inw, const float* inb,
                      const float* ow, const float* obias,
                      const Scratch& sc) {
    int QT = cdiv(Sq, 16);
    layernorm_bf16_kernel<<<Mq, 256, 0, st>>>(x, lnw, lnb1, sc.lnb, DM);
    f2bf_kernel<<<cdiv(Mk * DM, 256), 256, 0, st>>>(ctx, sc.ctxb, Mk * DM);
    // split (D, 3D) into Wq/Wk/Wv, each transposed -> bf16 N x K
    launch_transpose(st, inw, sc.wqkvT + 0 * DM * DM, DM, DM, 3 * DM, 0);
    launch_transpose(st, inw, sc.wqkvT + 1 * DM * DM, DM, DM, 3 * DM, DM);
    launch_transpose(st, inw, sc.wqkvT + 2 * DM * DM, DM, DM, 3 * DM, 2 * DM);
    launch_gemm(st, sc.lnb,  sc.wqkvT + 0 * DM * DM, inb + 0 * DM, nullptr, nullptr, sc.qb, Mq, DM, DM, 1, 0, Sq, Skp);
    launch_gemm(st, sc.ctxb, sc.wqkvT + 1 * DM * DM, inb + 1 * DM, nullptr, nullptr, sc.kb, Mk, DM, DM, 1, 0, Sk, Skp);
    if (Skp != Sk) {
        int nvt = BZ * NH * HD * Skp;
        zero_bf16_kernel<<<cdiv(nvt, 256), 256, 0, st>>>(sc.vtb, nvt);
    }
    launch_gemm(st, sc.ctxb, sc.wqkvT + 2 * DM * DM, inb + 2 * DM, nullptr, nullptr, sc.vtb, Mk, DM, DM, 2, 0, Sk, Skp);
    attn_kernel<<<(BZ * NH * QT) / 2, 64, 0, st>>>(sc.qb, sc.kb, sc.vtb, sc.ob, Sq, Sk, Skp, QT);
    launch_transpose(st, ow, sc.wT1, DM, DM, DM, 0);
    launch_gemm(st, sc.ob, sc.wT1, obias, x, x, nullptr, Mq, DM, DM, 0, 0, Sq, Skp);
}

extern "C" void kernel_launch(void* const* d_in, const int* in_sizes, int n_in,
                              void* d_out, int out_size, void* d_ws, size_t ws_size,
                              hipStream_t stream) {
    const float* x_spectra = (const float*)d_in[0];
    const float* x_gaia    = (const float*)d_in[1];
    const float* ts_w = (const float*)d_in[2];
    const float* ts_b = (const float*)d_in[3];
    const float* tg_w = (const float*)d_in[4];
    const float* tg_b = (const float*)d_in[5];
    const float* s_w[12]; for (int i = 0; i < 12; ++i) s_w[i] = (const float*)d_in[6 + i];
    const float* g_w[12]; for (int i = 0; i < 12; ++i) g_w[i] = (const float*)d_in[18 + i];
    const float* cs_w[6]; for (int i = 0; i < 6; ++i)  cs_w[i] = (const float*)d_in[30 + i];
    const float* cg_w[6]; for (int i = 0; i < 6; ++i)  cg_w[i] = (const float*)d_in[36 + i];
    const float* cls_lnw = (const float*)d_in[42];
    const float* cls_lnb = (const float*)d_in[43];
    const float* cls_w   = (const float*)d_in[44];
    const float* cls_b   = (const float*)d_in[45];
    float* out = (float*)d_out;

    // ---- workspace carve -------------------------------------------------
    char* wp = (char*)d_ws;
    auto carve = [&](size_t bytes) -> void* {
        void* p = (void*)wp;
        wp += (bytes + 255) & ~(size_t)255;
        return p;
    };
    Scratch sc;
    sc.xs    = (float*) carve((size_t)MS * DM * 4);
    sc.xg    = (float*) carve((size_t)MG * DM * 4);
    sc.lnb   = (ushort*)carve((size_t)MS * DM * 2);
    sc.qb    = (ushort*)carve((size_t)MS * DM * 2);
    sc.kb    = (ushort*)carve((size_t)MS * DM * 2);
    sc.vtb   = (ushort*)carve((size_t)MS * DM * 2);   // covers B*H*HD*256
    sc.ob    = (ushort*)carve((size_t)MS * DM * 2);
    sc.hb    = (ushort*)carve((size_t)MS * 4 * DM * 2);
    sc.ctxb  = (ushort*)carve((size_t)MS * DM * 2);
    sc.wqkvT = (ushort*)carve((size_t)3 * DM * DM * 2);
    sc.wT1   = (ushort*)carve((size_t)4 * DM * DM * 2);
    sc.wT2   = (ushort*)carve((size_t)4 * DM * DM * 2);
    sc.fused = (float*) carve((size_t)BZ * 2 * DM * 4);

    // ---- tokenize --------------------------------------------------------
    token_kernel<<<cdiv(MS * DM, 256), 256, 0, stream>>>(x_spectra, ts_w, ts_b, sc.xs, 64, DM, MS * DM);
    token_kernel<<<cdiv(MG * DM, 256), 256, 0, stream>>>(x_gaia,    tg_w, tg_b, sc.xg, 2,  DM, MG * DM);

    // ---- branches --------------------------------------------------------
    run_branch(stream, sc.xs, MS, SS, 256,
               s_w[0], s_w[1], s_w[2], s_w[3], s_w[4], s_w[5],
               s_w[6], s_w[7], s_w[8], s_w[9], s_w[10], s_w[11], sc);
    run_branch(stream, sc.xg, MG, SG, 32,
               g_w[0], g_w[1], g_w[2], g_w[3], g_w[4], g_w[5],
               g_w[6], g_w[7], g_w[8], g_w[9], g_w[10], g_w[11], sc);

    // ---- cross attention (sequential: cg sees updated xs) ----------------
    run_cross(stream, sc.xs, MS, SS, sc.xg, MG, SG, 32,
              cs_w[0], cs_w[1], cs_w[2], cs_w[3], cs_w[4], cs_w[5], sc);
    run_cross(stream, sc.xg, MG, SG, sc.xs, MS, SS, 256,
              cg_w[0], cg_w[1], cg_w[2], cg_w[3], cg_w[4], cg_w[5], sc);

    // ---- pool + classifier ----------------------------------------------
    pool_kernel<<<cdiv(BZ * 2 * DM, 256), 256, 0, stream>>>(sc.xs, sc.xg, sc.fused);
    cls_kernel<<<BZ, 256, 0, stream>>>(sc.fused, cls_lnw, cls_lnb, cls_w, cls_b, out);
}
